// TrilinearFusion_7086696038813
// MI455X (gfx1250) — 8-bit matrix operand order repaired and checked by reading, compiled, not run on hardware
//
#include <hip/hip_runtime.h>

#define BATCH 4096
#define SD 32
#define SP1 33
#define IJ 1089
#define IJP 1152
#define NS64 18
#define NK 33
#define NT 6
#define HID 96
#define KSEG 4

typedef int   v8i __attribute__((ext_vector_type(8)));
typedef float v8f __attribute__((ext_vector_type(8)));

union FragF8 { uint4 u[2]; v8i v; };

__device__ __forceinline__ unsigned char f32_to_e4m3(float f)
{
    unsigned int s = (__float_as_uint(f) >> 24) & 0x80;
    float af = fabsf(f);
    if (!(af < 448.f)) af = 448.f;
    int e = (int)((__float_as_uint(af) >> 23) & 0xff) - 127;
    if (af == 0.f || e < -10) return (unsigned char)s;
    if (e < -6) e = -6;
    float scale = __uint_as_float((unsigned int)(127 + 3 - e) << 23);
    int mant = (int)rintf(af * scale);
    if (mant >= 16) { mant >>= 1; ++e; }
    if (mant < 8) return (unsigned char)(s | mant);
    int be = e + 7;
    if (be > 15) return (unsigned char)(s | 0x7e);
    return (unsigned char)(s | (be << 3) | (mant - 8));
}

__global__ __launch_bounds__(256)
void gate_kernel(const float* __restrict__ x,  const float* __restrict__ xg,
                 const float* __restrict__ rw, const float* __restrict__ rb,
                 const float* __restrict__ gw, const float* __restrict__ gb,
                 const float* __restrict__ ow, const float* __restrict__ ob,
                 float* __restrict__ out)
{
    const int o    = threadIdx.x & 31;
    const int wid  = threadIdx.x >> 5;
    const int wr0  = wid * 8;
    const int row0 = blockIdx.x * 64;

    __shared__ float sx[64][32];
    __shared__ float sv[64][32];
    __shared__ float sg[64][32];

#pragma unroll
    for (int r = 0; r < 8; ++r) {
        sx[wr0 + r][o] = x [(row0 + wr0 + r) * SD + o];
        sv[wr0 + r][o] = xg[(row0 + wr0 + r) * SD + o];
    }
    __syncthreads();

    float acc[8];
#pragma unroll
    for (int r = 0; r < 8; ++r) acc[r] = gb[o];

    const float4* gwo = (const float4*)(gw + o * 1024);
    for (int q = 0; q < 256; ++q) {
        float4 g4 = gwo[q];
#pragma unroll
        for (int c = 0; c < 4; ++c) {
            int   ij = q * 4 + c;
            float g  = (&g4.x)[c];
            int   i  = ij >> 5, j = ij & 31;
#pragma unroll
            for (int r = 0; r < 8; ++r)
                acc[r] += g * (sx[wr0 + r][i] * sv[wr0 + r][j]);
        }
    }

#pragma unroll
    for (int r = 0; r < 8; ++r) {
        float a = rb[o];
        for (int i = 0; i < SD; ++i) a += sx[wr0 + r][i] * rw[i * SD + o];
        float rl = fmaxf(a, 0.f);
        float w  = 1.f / (1.f + __expf(-acc[r]));
        sg[wr0 + r][o] = w * rl;
    }
    __syncthreads();

#pragma unroll
    for (int r = 0; r < 8; ++r) {
        float a = ob[o];
        for (int i = 0; i < SD; ++i) a += sg[wr0 + r][i] * ow[i * SD + o];
        int row = row0 + wr0 + r;
        out[row * SP1 + o] = fmaxf(a, 0.f);
        if (o == 0) out[row * SP1 + 32] = 1.0f;
    }
}

#define W1P_TOT (NK * NS64 * NT * 1024)
__global__ __launch_bounds__(256)
void repack_kernel(const float* __restrict__ w1, unsigned char* __restrict__ w1p)
{
    int id = blockIdx.x * 256 + threadIdx.x;
    if (id >= W1P_TOT) return;
    int w  = id & 31;
    int L  = (id >> 5) & 31;
    int g  = id >> 10;
    int t  = g % 6; g /= 6;
    int s  = g % NS64;
    int k  = g / NS64;
    int hiL = L >> 4;
    int nl  = L & 15;
    int K   = (w >> 3) * 16 + hiL * 8 + (w & 7);
    int ij  = s * 64 + K;
    float val = 0.f;
    if (ij < IJ) val = 64.f * w1[(size_t)(ij * NK + k) * HID + t * 16 + nl];
    w1p[id] = f32_to_e4m3(val);
}

__global__ __launch_bounds__(32)
void trilinear_wmma_kernel(const float* __restrict__ o1, const float* __restrict__ o2,
                           const float* __restrict__ o3, const unsigned char* __restrict__ w1p,
                           float* __restrict__ partial)
{
    const int lane    = threadIdx.x;
    const int rowBase = blockIdx.x * 16;
    const int seg     = blockIdx.y;
    const int k0      = (seg * NK) / KSEG;
    const int k1      = ((seg + 1) * NK) / KSEG;

    __shared__ __align__(16) unsigned char u12s[16 * IJP];
    __shared__ float o3s[16 * NK];

    for (int idx = lane; idx < 16 * IJP; idx += 32) {
        int r = idx / IJP, ij = idx - r * IJP;
        float v = 0.f;
        if (ij < IJ) {
            int i = ij / SP1, j = ij - i * SP1;
            v = 16.f * o1[(rowBase + r) * SP1 + i] * o2[(rowBase + r) * SP1 + j];
        }
        int s  = ij >> 6;
        int kk = ij & 63;
        int p  = ((kk >> 3) & 1) * 32 + ((kk >> 4) & 3) * 8 + (kk & 7);
        u12s[(r * NS64 + s) * 64 + p] = f32_to_e4m3(v);
    }
    for (int idx = lane; idx < 16 * NK; idx += 32) {
        int r = idx / NK, k = idx - r * NK;
        o3s[idx] = o3[(rowBase + r) * SP1 + k];
    }
    __syncthreads();

    const int mrow = lane & 15;
    const int hi   = lane >> 4;
    const unsigned char* aRow = &u12s[mrow * NS64 * 64 + hi * 32];

    float fin[NT][8] = {};

    for (int k = k0; k < k1; ++k) {
        v8f acc[NT] = {};
        const unsigned char* bk = w1p + (size_t)k * NS64 * NT * 1024;
        for (int s = 0; s < NS64; ++s) {
            FragF8 a;
            const uint4* ap = (const uint4*)(aRow + s * 64);
            a.u[0] = ap[0];
            a.u[1] = ap[1];
            const unsigned char* bs = bk + (size_t)s * NT * 1024;
#pragma unroll
            for (int t = 0; t < NT; ++t) {
                FragF8 b;
                const uint4* bp = (const uint4*)(bs + t * 1024 + lane * 32);
                b.u[0] = bp[0];
                b.u[1] = bp[1];
                acc[t] = __builtin_amdgcn_wmma_f32_16x16x64_fp8_fp8(
                    a.v, b.v, (short)0, acc[t], false, false);
            }
        }
#pragma unroll
        for (int r = 0; r < 8; ++r) {
            float ov = o3s[(hi * 8 + r) * NK + k] * (1.0f / 1024.0f);
#pragma unroll
            for (int t = 0; t < NT; ++t) fin[t][r] += ov * acc[t][r];
        }
    }

    float* pseg = partial + (size_t)seg * BATCH * HID;
    const int nl = lane & 15;
#pragma unroll
    for (int t = 0; t < NT; ++t)
#pragma unroll
        for (int r = 0; r < 8; ++r)
            pseg[(size_t)(rowBase + hi * 8 + r) * HID + t * 16 + nl] = fin[t][r];
}

__global__ __launch_bounds__(256)
void reduce_relu_kernel(const float* __restrict__ partial, const float* __restrict__ b1,
                        float* __restrict__ H)
{
    int id = blockIdx.x * 256 + threadIdx.x;
    if (id >= BATCH * HID) return;
    float s = b1[id % HID];
#pragma unroll
    for (int g = 0; g < KSEG; ++g) s += partial[(size_t)g * BATCH * HID + id];
    H[id] = fmaxf(s, 0.f);
}

__global__ __launch_bounds__(256)
void enc2_kernel(const float* __restrict__ H, const float* __restrict__ w2,
                 const float* __restrict__ b2, float* __restrict__ out)
{
    int id = blockIdx.x * 256 + threadIdx.x;
    if (id >= BATCH * HID) return;
    int row = id / HID, n = id - row * HID;
    float s = b2[n];
    const float* h = H + row * HID;
    for (int m = 0; m < HID; ++m) s += h[m] * w2[m * HID + n];
    out[id] = fmaxf(s, 0.f);
}

extern "C" void kernel_launch(void* const* d_in, const int* in_sizes, int n_in,
                              void* d_out, int out_size, void* d_ws, size_t ws_size,
                              hipStream_t stream)
{
    (void)in_sizes; (void)n_in; (void)out_size; (void)ws_size;

    const float* x_path  = (const float*)d_in[0];
    const float* x_graph = (const float*)d_in[1];
    const float* x_omic  = (const float*)d_in[2];
    const float* enc1_w  = (const float*)d_in[21];
    const float* enc1_b  = (const float*)d_in[22];
    const float* enc2_w  = (const float*)d_in[23];
    const float* enc2_b  = (const float*)d_in[24];
    float* out = (float*)d_out;

    char* ws = (char*)d_ws;
    float*         o1      = (float*)(ws + 0);
    float*         o2      = (float*)(ws + 540672);
    float*         o3      = (float*)(ws + 1081344);
    unsigned char* w1p     = (unsigned char*)(ws + 1622016);
    float*         partial = (float*)(ws + 5271552);
    float*         H       = (float*)(ws + 5271552 + (size_t)KSEG * BATCH * HID * 4);

    gate_kernel<<<BATCH / 64, 256, 0, stream>>>(x_path, x_omic,
        (const float*)d_in[3], (const float*)d_in[4], (const float*)d_in[5],
        (const float*)d_in[6], (const float*)d_in[7], (const float*)d_in[8], o1);
    gate_kernel<<<BATCH / 64, 256, 0, stream>>>(x_graph, x_omic,
        (const float*)d_in[9], (const float*)d_in[10], (const float*)d_in[11],
        (const float*)d_in[12], (const float*)d_in[13], (const float*)d_in[14], o2);
    gate_kernel<<<BATCH / 64, 256, 0, stream>>>(x_path, x_omic,
        (const float*)d_in[15], (const float*)d_in[16], (const float*)d_in[17],
        (const float*)d_in[18], (const float*)d_in[19], (const float*)d_in[20], o3);

    repack_kernel<<<(W1P_TOT + 255) / 256, 256, 0, stream>>>(enc1_w, w1p);

    trilinear_wmma_kernel<<<dim3(BATCH / 16, KSEG), 32, 0, stream>>>(o1, o2, o3, w1p, partial);

    reduce_relu_kernel<<<(BATCH * HID + 255) / 256, 256, 0, stream>>>(partial, enc1_b, H);

    enc2_kernel<<<(BATCH * HID + 255) / 256, 256, 0, stream>>>(H, enc2_w, enc2_b, out);
}
